// SubCentroidsHead_53815940219467
// MI455X (gfx1250) — compile-verified, matrix operand order checked by reading, not run on hardware
//
#include <hip/hip_runtime.h>
#include <hip/hip_bf16.h>

// ---------------- problem constants ----------------
#define N_ROWS   131072
#define DIM      768
#define KCLS     100
#define KM       400        // K*M prototype rows
#define KMP      416        // padded to multiple of 16
#define TILE_ROWS 64        // x rows per block
#define KSTEP    32         // WMMA K per step (f16)
#define NSTEPS   (DIM / KSTEP)      // 24
#define CTILES   (KMP / 16)         // 26 column tiles
#define CT_PER_WAVE 13              // 26 tiles split across 2 wave-groups
#define XS_STRIDE 776               // 768 + 8 halves pad (conflict-free ds_load_b128)
#define BS_ROW_HALF 40              // 32 data halves + 8 pad halves = 80 B / row (TDM pad)
#define BS_BUF_BYTES (KMP * BS_ROW_HALF * 2)   // 33,280 B per buffer
#define SM_STRIDE 104               // 100 group-max cols padded

typedef __attribute__((ext_vector_type(4)))  float     f32x4;
typedef __attribute__((ext_vector_type(8)))  float     v8f;
typedef __attribute__((ext_vector_type(4)))  _Float16  v4h;
typedef __attribute__((ext_vector_type(8)))  _Float16  v8h;
typedef __attribute__((ext_vector_type(16))) _Float16  v16h;
typedef __attribute__((ext_vector_type(4)))  unsigned int u32x4;
typedef __attribute__((ext_vector_type(8)))  int       i32x8;
typedef __attribute__((ext_vector_type(4)))  int       i32x4;

__device__ __forceinline__ float wave_sum(float v) {
#pragma unroll
    for (int o = 16; o > 0; o >>= 1) v += __shfl_xor(v, o, 32);
    return v;
}

// ---------------------------------------------------------------------------
// Kernel 1: L2-normalize prototype rows -> f16, zero the 16 pad rows.
// ---------------------------------------------------------------------------
__global__ void proto_norm_kernel(const float* __restrict__ P,
                                  _Float16* __restrict__ P16) {
    __shared__ float red[8];
    const int row = blockIdx.x;
    const int t = threadIdx.x;
    if (row >= KM) {                                 // pad rows -> zeros
        for (int d = t; d < DIM; d += 256) P16[(size_t)row * DIM + d] = (_Float16)0.f;
        return;
    }
    const float* pr = P + (size_t)row * DIM;
    float v0 = pr[t], v1 = pr[t + 256], v2 = pr[t + 512];
    float ss = v0 * v0 + v1 * v1 + v2 * v2;
    ss = wave_sum(ss);
    if ((t & 31) == 0) red[t >> 5] = ss;
    __syncthreads();
    float tot = 0.f;
#pragma unroll
    for (int i = 0; i < 8; ++i) tot += red[i];
    float sc = 1.0f / fmaxf(sqrtf(tot), 1e-12f);
    _Float16* o = P16 + (size_t)row * DIM;
    o[t]       = (_Float16)(v0 * sc);
    o[t + 256] = (_Float16)(v1 * sc);
    o[t + 512] = (_Float16)(v2 * sc);
}

// ---------------------------------------------------------------------------
// TDM: DMA one B k-slice  (KMP rows x 32 halves, row stride DIM halves)
// into LDS with 4-dword padding after every 16 dwords (64B row -> 80B row).
// 6-arg builtin form (clang-23 / therock headers).
// ---------------------------------------------------------------------------
__device__ __forceinline__ void tdm_load_b(const _Float16* P16, int kk, unsigned lds_off) {
    unsigned long long ga = (unsigned long long)(uintptr_t)(P16 + kk);
    u32x4 g0;
    g0[0] = 1u;                                         // count=1, user descriptor
    g0[1] = lds_off;                                    // LDS byte address
    g0[2] = (unsigned)(ga & 0xFFFFFFFFu);               // global_addr[31:0]
    g0[3] = (unsigned)((ga >> 32) & 0x1FFFFFFu)         // global_addr[56:32]
          | (2u << 30);                                 // type = 2 ("image")
    i32x8 g1;
    g1[0] = (int)((1u << 16)      // data_size = 2 bytes
                | (1u << 20)      // pad_enable
                | (3u << 22)      // pad_interval: 16 dwords (64B)
                | (3u << 25));    // pad_amount: 4 dwords (16B) -> 80B row stride
    g1[1] = (int)(((unsigned)DIM & 0xFFFFu) << 16);     // tensor_dim0[15:0]
    g1[2] = (int)(((unsigned)DIM >> 16) | (((unsigned)KMP & 0xFFFFu) << 16)); // dim0 hi | dim1 lo
    g1[3] = (int)(((unsigned)KMP >> 16) | ((unsigned)KSTEP << 16));           // dim1 hi | tile_dim0=32
    g1[4] = (int)((unsigned)KMP);                       // tile_dim1=416, tile_dim2=0
    g1[5] = (int)DIM;                                   // tensor_dim0_stride low
    g1[6] = 0;
    g1[7] = 0;
    i32x4 z4 = {0, 0, 0, 0};
    i32x8 z8 = {0, 0, 0, 0, 0, 0, 0, 0};
    __builtin_amdgcn_tensor_load_to_lds(g0, g1, z4, z4, z8, 0);
}

// ---------------------------------------------------------------------------
// Fused: layernorm + L2 (x)  ->  f16 WMMA GEMM vs protos  ->  max over M=4
//        ->  layernorm over K=100  ->  out
// ---------------------------------------------------------------------------
__global__ __launch_bounds__(256)
void subcentroid_fused_kernel(const float* __restrict__ x,
                              const _Float16* __restrict__ P16,
                              const float* __restrict__ fg,
                              const float* __restrict__ fb,
                              const float* __restrict__ mg,
                              const float* __restrict__ mb,
                              float* __restrict__ out) {
    __shared__ __align__(16) unsigned char smem[TILE_ROWS * XS_STRIDE * 2 + 2 * BS_BUF_BYTES];
    _Float16* Xs  = (_Float16*)smem;
    _Float16* Bs0 = (_Float16*)(smem + TILE_ROWS * XS_STRIDE * 2);
    _Float16* Bs1 = (_Float16*)(smem + TILE_ROWS * XS_STRIDE * 2 + BS_BUF_BYTES);
    float*    Sm  = (float*)Bs0;   // aliased after GEMM: 64 x 104 f32 = 26.6KB < 33.3KB

    const int t    = threadIdx.x;
    const int lane = t & 31;
    const int wave = __builtin_amdgcn_readfirstlane(t >> 5);   // SGPR -> s_cbranch gating
    const int row0 = blockIdx.x * TILE_ROWS;

    // Kick off TDM for k-slice 0 immediately; overlaps with phase-1 math.
    if (wave == 0) tdm_load_b(P16, 0, (unsigned)(uintptr_t)(void*)Bs0);

    // ---------- Phase 1: per-row layernorm + L2-normalize -> Xs (f16) ----------
    float g[24], bt[24];
#pragma unroll
    for (int c = 0; c < 6; ++c) {
        f32x4 gv = *(const f32x4*)(fg + c * 128 + lane * 4);
        f32x4 bv = *(const f32x4*)(fb + c * 128 + lane * 4);
#pragma unroll
        for (int j = 0; j < 4; ++j) { g[c * 4 + j] = gv[j]; bt[c * 4 + j] = bv[j]; }
    }
#pragma unroll 1
    for (int r8 = 0; r8 < 8; ++r8) {
        const int row = wave * 8 + r8;
        const float* xr = x + (size_t)(row0 + row) * DIM;
        float xv[24], s = 0.f, ss = 0.f;
#pragma unroll
        for (int c = 0; c < 6; ++c) {
            f32x4 v4 = *(const f32x4*)(xr + c * 128 + lane * 4);
#pragma unroll
            for (int j = 0; j < 4; ++j) {
                float v = v4[j]; xv[c * 4 + j] = v; s += v; ss += v * v;
            }
        }
        s = wave_sum(s); ss = wave_sum(ss);
        const float mu   = s * (1.f / DIM);
        const float var  = ss * (1.f / DIM) - mu * mu;
        const float rstd = rsqrtf(var + 1e-5f);
        float l2 = 0.f;
#pragma unroll
        for (int i = 0; i < 24; ++i) {
            float v = (xv[i] - mu) * rstd * g[i] + bt[i];
            xv[i] = v; l2 += v * v;
        }
        l2 = wave_sum(l2);
        const float sc = 1.f / fmaxf(sqrtf(l2), 1e-12f);
#pragma unroll
        for (int c = 0; c < 6; ++c) {
            v4h h;
#pragma unroll
            for (int j = 0; j < 4; ++j) h[j] = (_Float16)(xv[c * 4 + j] * sc);
            *(v4h*)(Xs + row * XS_STRIDE + c * 128 + lane * 4) = h;
        }
    }
    __syncthreads();

    // ---------- Phase 2: WMMA GEMM, TDM double-buffered over K ----------
    const int rt      = wave & 3;                 // row tile 0..3
    const int cbase   = (wave >> 2) * CT_PER_WAVE;
    const int halfsel = (lane >> 4) * 8;          // K sub-select per half-wave
    const _Float16* ap = Xs + (rt * 16 + (lane & 15)) * XS_STRIDE + halfsel;
    const int bcol = (lane & 15);

    v8f acc[CT_PER_WAVE] = {};

#pragma unroll 1
    for (int ks = 0; ks < NSTEPS; ++ks) {
        if (wave == 0) __builtin_amdgcn_s_wait_tensorcnt(0);   // cur buffer landed
        __syncthreads();
        const _Float16* cur = (ks & 1) ? Bs1 : Bs0;
        _Float16*       nxt = (ks & 1) ? Bs0 : Bs1;
        if (wave == 0 && ks + 1 < NSTEPS)
            tdm_load_b(P16, (ks + 1) * KSTEP, (unsigned)(uintptr_t)(void*)nxt);

        // A fragment (shared by all 13 column tiles this wave owns)
        v8h alo = *(const v8h*)(ap + ks * KSTEP);
        v8h ahi = *(const v8h*)(ap + ks * KSTEP + 16);
        v16h a;
#pragma unroll
        for (int i = 0; i < 8; ++i) { a[i] = alo[i]; a[8 + i] = ahi[i]; }

        // B fragments software-pipelined at distance 2: bfrag[j&1] is refilled
        // only after WMMA j consumed it and is next read by WMMA j+2, so two
        // fragments are simultaneously live -> allocator cannot coalesce them,
        // and each wait covers loads issued two WMMAs earlier.
        auto loadB = [&](int j) -> v16h {
            const _Float16* bp = cur + ((cbase + j) * 16 + bcol) * BS_ROW_HALF + halfsel;
            v8h lo = *(const v8h*)bp;
            v8h hi = *(const v8h*)(bp + 16);
            v16h b;
#pragma unroll
            for (int i = 0; i < 8; ++i) { b[i] = lo[i]; b[8 + i] = hi[i]; }
            return b;
        };
        v16h bfrag[2];
        bfrag[0] = loadB(0);
        bfrag[1] = loadB(1);
#pragma unroll
        for (int j = 0; j < CT_PER_WAVE; ++j) {
            acc[j] = __builtin_amdgcn_wmma_f32_16x16x32_f16(
                false, a, false, bfrag[j & 1], (short)0, acc[j], false, false);
            if (j + 2 < CT_PER_WAVE) bfrag[j & 1] = loadB(j + 2);
        }
    }
    __syncthreads();   // everyone done with B buffers before Sm aliases them

    // ---------- Phase 3: max over M=4 (adjacent cols = adjacent lanes) ----------
    const int mrowb = rt * 16 + 8 * (lane >> 4);
#pragma unroll
    for (int j = 0; j < CT_PER_WAVE; ++j) {
        const int ct = cbase + j;
#pragma unroll
        for (int v = 0; v < 8; ++v) {
            float m = acc[j][v];
            m = fmaxf(m, __shfl_xor(m, 1, 32));
            m = fmaxf(m, __shfl_xor(m, 2, 32));
            if ((lane & 3) == 0) {
                const int kcol = ct * 4 + ((lane & 15) >> 2);
                Sm[(mrowb + v) * SM_STRIDE + kcol] = m;
            }
        }
    }
    __syncthreads();

    // ---------- Phase 4: layernorm over K=100 per row, write out ----------
#pragma unroll 1
    for (int r8 = 0; r8 < 8; ++r8) {
        const int row = wave * 8 + r8;
        float vals[4], s = 0.f, ss = 0.f;
#pragma unroll
        for (int q = 0; q < 4; ++q) {
            const int c = lane + q * 32;
            float v = (c < KCLS) ? Sm[row * SM_STRIDE + c] : 0.f;
            vals[q] = v; s += v; ss += v * v;
        }
        s = wave_sum(s); ss = wave_sum(ss);
        const float mu   = s * (1.f / KCLS);
        const float var  = ss * (1.f / KCLS) - mu * mu;
        const float rstd = rsqrtf(var + 1e-5f);
        float* orow = out + (size_t)(row0 + row) * KCLS;
#pragma unroll
        for (int q = 0; q < 4; ++q) {
            const int c = lane + q * 32;
            if (c < KCLS) orow[c] = (vals[q] - mu) * rstd * mg[c] + mb[c];
        }
    }
}

// ---------------------------------------------------------------------------
extern "C" void kernel_launch(void* const* d_in, const int* in_sizes, int n_in,
                              void* d_out, int out_size, void* d_ws, size_t ws_size,
                              hipStream_t stream) {
    const float* x  = (const float*)d_in[0];
    const float* P  = (const float*)d_in[1];
    const float* fg = (const float*)d_in[2];
    const float* fb = (const float*)d_in[3];
    const float* mg = (const float*)d_in[4];
    const float* mb = (const float*)d_in[5];
    float* out = (float*)d_out;
    _Float16* P16 = (_Float16*)d_ws;   // 416*768*2 = 638,976 B

    proto_norm_kernel<<<KMP, 256, 0, stream>>>(P, P16);
    subcentroid_fused_kernel<<<N_ROWS / TILE_ROWS, 256, 0, stream>>>(
        x, P16, fg, fb, mg, mb, out);
}
